// InteractionBlock_53721450938959
// MI455X (gfx1250) — matrix operand order repaired and checked by reading, not compiled, not run on hardware
//
#include <hip/hip_runtime.h>
#include <hip/hip_bf16.h>
#include <math.h>

// ---------------- types ----------------
typedef __attribute__((ext_vector_type(16))) __bf16 v16bf;
typedef __attribute__((ext_vector_type(8)))  float  v8f;

// Pack two f32 -> two bf16 (truncation) in one v_perm_b32.
static __device__ __forceinline__ unsigned int pack_bf16x2(float lo, float hi) {
    union { float f; unsigned int u; } a, b;
    a.f = lo; b.f = hi;
#if __has_builtin(__builtin_amdgcn_perm)
    // result bytes: [1:0]=lo[3:2], [3:2]=hi[3:2]
    return __builtin_amdgcn_perm(b.u, a.u, 0x07060302u);
#else
    return (a.u >> 16) | (b.u & 0xffff0000u);
#endif
}

// ---------------- generic bf16 WMMA GEMM with fused epilogue ----------------
// C[M,N] = epilogue( A[M,K] @ W[K,N] + bias )
//   MODE 0: C = AB + bias
//   MODE 1: C = resid + gamma[col] * (AB + bias)
//   MODE 2: C = resid + (AB + bias)
// grid = (N/(16*NTILES), ceil(M/128)); block = 256 threads (8 waves).
// Each wave owns a 16-row stripe and NTILES 16x16 output tiles; the block's
// (16*NTILES)-wide W stripe is staged once per 64-deep K chunk in LDS as bf16,
// and one A fragment feeds NTILES back-to-back WMMAs.
// Requires M%16==0, N%(16*NTILES)==0, K%64==0.
template <int MODE, int NTILES>
__global__ void gemm_bf16_wmma(const float* __restrict__ A, const float* __restrict__ W,
                               const float* __restrict__ bias, const float* __restrict__ resid,
                               const float* __restrict__ gamma, float* __restrict__ C,
                               int M, int N, int K)
{
    constexpr int NCOLS = NTILES * 16;
    __shared__ unsigned short sW[NCOLS * 64];    // [col][k], 64-deep K chunk

    const int lane = threadIdx.x & 31;
    const int wave = threadIdx.x >> 5;
    const int rowBase = blockIdx.y * 128 + wave * 16;
    const int colBase = blockIdx.x * NCOLS;
    const int r  = lane & 15;
    const int hi = lane >> 4;
    const bool active = (rowBase < M);           // wave-uniform

    const float* __restrict__ arow = A + (long long)(rowBase + r) * K;
    v8f acc[NTILES];
#pragma unroll
    for (int t = 0; t < NTILES; ++t) acc[t] = {};

    for (int k0 = 0; k0 < K; k0 += 64) {
        __syncthreads();
        // ---- cooperative fill: NCOLS cols x 64 k of W -> bf16 (truncated) in LDS ----
#pragma unroll
        for (int e = 0; e < NCOLS / 4; ++e) {
            const int idx = threadIdx.x + e * 256;       // 0 .. NCOLS*64-1
            const int col = idx % NCOLS;                 // consecutive tid -> consecutive col
            const int k   = idx / NCOLS;
            union { float f; unsigned int u; } cv;
            cv.f = W[(long long)(k0 + k) * N + colBase + col];
            sW[col * 64 + k] = (unsigned short)(cv.u >> 16);
        }
        __syncthreads();

        if (active) {
#pragma unroll
            for (int s = 0; s < 2; ++s) {        // two K=32 sub-chunks
                const int kk = k0 + s * 32;
                // ---- A fragment: 16x32 bf16, ISA lane layout ----
                union { v16bf v; unsigned int u32[8]; } af;
                const float4* p0 = (const float4*)(arow + kk + hi * 8);
                const float4* p1 = (const float4*)(arow + kk + 16 + hi * 8);
                float4 a0 = p0[0], a1 = p0[1], a2 = p1[0], a3 = p1[1];
                af.u32[0] = pack_bf16x2(a0.x, a0.y);
                af.u32[1] = pack_bf16x2(a0.z, a0.w);
                af.u32[2] = pack_bf16x2(a1.x, a1.y);
                af.u32[3] = pack_bf16x2(a1.z, a1.w);
                af.u32[4] = pack_bf16x2(a2.x, a2.y);
                af.u32[5] = pack_bf16x2(a2.z, a2.w);
                af.u32[6] = pack_bf16x2(a3.x, a3.y);
                af.u32[7] = pack_bf16x2(a3.z, a3.w);

                // ---- one A fragment feeds NTILES WMMAs ----
#pragma unroll
                for (int t = 0; t < NTILES; ++t) {
                    union { v16bf v; uint4 q[2]; } bfm;
                    const uint4* bp =
                        (const uint4*)(sW + (t * 16 + r) * 64 + s * 32 + hi * 8);
                    bfm.q[0] = bp[0];
                    bfm.q[1] = bp[2];
                    acc[t] = __builtin_amdgcn_wmma_f32_16x16x32_bf16(
                        false, af.v, false, bfm.v, (short)0, acc[t], false, false);
                }
            }
        }
    }

    if (!active) return;

    // ---- epilogue (branch-free: MODE is compile-time) ----
#pragma unroll
    for (int t = 0; t < NTILES; ++t) {
        const int col = colBase + t * 16 + r;
        const float bv = bias[col];
        float gv = 0.0f;
        if (MODE == 1) gv = gamma[col];
#pragma unroll
        for (int v = 0; v < 8; ++v) {
            long long idx = (long long)(rowBase + hi * 8 + v) * N + col;
            float val = acc[t][v] + bv;
            if (MODE == 1)      val = resid[idx] + gv * val;
            else if (MODE == 2) val = resid[idx] + val;
            C[idx] = val;
        }
    }
}

// ---------------- LayerNorm: one wave per 768-wide row ----------------
__global__ void layernorm_kernel(const float* __restrict__ X, const float* __restrict__ g,
                                 const float* __restrict__ b, float* __restrict__ Y,
                                 long long rows)
{
    const int wave = threadIdx.x >> 5;
    const int lane = threadIdx.x & 31;
    long long row = (long long)blockIdx.x * 8 + wave;
    if (row >= rows) return;
    const float* __restrict__ xr = X + row * 768;
    float vals[24];
    float s = 0.f, s2 = 0.f;
#pragma unroll
    for (int i = 0; i < 24; ++i) {
        float v = xr[lane + i * 32];
        vals[i] = v; s += v; s2 += v * v;
    }
#pragma unroll
    for (int o = 16; o > 0; o >>= 1) {
        s  += __shfl_xor(s,  o, 32);
        s2 += __shfl_xor(s2, o, 32);
    }
    const float mu   = s * (1.0f / 768.0f);
    const float var  = s2 * (1.0f / 768.0f) - mu * mu;
    const float rstd = rsqrtf(var + 1e-6f);
    float* __restrict__ yr = Y + row * 768;
#pragma unroll
    for (int i = 0; i < 24; ++i) {
        int c = lane + i * 32;
        yr[c] = (vals[i] - mu) * rstd * g[c] + b[c];
    }
}

// ---------------- multi-scale deformable attention sampling ----------------
struct LevelsDesc { int L; int h[4]; int w[4]; int start[4]; };

// value: (B, Nf, 768) row-major (head h occupies cols h*64..h*64+63)
// ref:   (B, Nq, L, 2) ; off: (B,Nq,12,L,4,2) ; awl: (B,Nq,12,L*4) logits
// out:   (B, Nq, 768)
__global__ void msda_sample_kernel(const float* __restrict__ value, const float* __restrict__ ref,
                                   const float* __restrict__ off, const float* __restrict__ awl,
                                   float* __restrict__ out, int Nq, int Nf, LevelsDesc lv)
{
    long long id = (long long)blockIdx.x * blockDim.x + threadIdx.x;
    long long total = 2LL * Nq * 768;
    if (id >= total) return;
    const int d  = (int)(id & 63);
    const int h  = (int)((id >> 6) % 12);
    long long bq = id / 768;
    const int q  = (int)(bq % Nq);
    const int b  = (int)(bq / Nq);
    const int LP = lv.L * 4;

    const float* __restrict__ awp  = awl + ((long long)(b * (long long)Nq + q) * 12 + h) * LP;
    const float* __restrict__ offp = off + ((long long)(b * (long long)Nq + q) * 12 + h) * (LP * 2);

    // softmax over L*P attention logits (shared by the 64 lanes of this head -> cached)
    float m = -1e30f;
    for (int i = 0; i < LP; ++i) m = fmaxf(m, awp[i]);
    float ssum = 0.f;
    for (int i = 0; i < LP; ++i) ssum += __expf(awp[i] - m);
    const float inv = 1.0f / ssum;

    const int ch = h * 64 + d;
    float acc = 0.f;
    for (int l = 0; l < lv.L; ++l) {
        const int hl = lv.h[l], wl = lv.w[l], st = lv.start[l];
        const float rx = ref[(((long long)b * Nq + q) * lv.L + l) * 2 + 0];
        const float ry = ref[(((long long)b * Nq + q) * lv.L + l) * 2 + 1];
        const float* __restrict__ vb = value + (long long)(b * (long long)Nf + st) * 768 + ch;
        for (int p = 0; p < 4; ++p) {
            const float ox = offp[(l * 4 + p) * 2 + 0];
            const float oy = offp[(l * 4 + p) * 2 + 1];
            const float wgt = __expf(awp[l * 4 + p] - m) * inv;
            const float lx = rx + ox / (float)wl;
            const float ly = ry + oy / (float)hl;
            const float px = lx * (float)wl - 0.5f;
            const float py = ly * (float)hl - 0.5f;
            const float x0 = floorf(px), y0 = floorf(py);
            const float dx = px - x0,   dy = py - y0;
            float sum = 0.f;
#pragma unroll
            for (int cy = 0; cy < 2; ++cy) {
#pragma unroll
                for (int cx = 0; cx < 2; ++cx) {
                    const float ix = x0 + cx, iy = y0 + cy;
                    const float cw = (cx ? dx : 1.f - dx) * (cy ? dy : 1.f - dy);
                    const bool valid = (ix >= 0.f) && (ix <= (float)(wl - 1)) &&
                                       (iy >= 0.f) && (iy <= (float)(hl - 1));
                    int xi = (int)fminf(fmaxf(ix, 0.f), (float)(wl - 1));
                    int yi = (int)fminf(fmaxf(iy, 0.f), (float)(hl - 1));
                    float v = vb[(long long)(yi * wl + xi) * 768];
                    sum += v * (valid ? cw : 0.f);
                }
            }
            acc += wgt * sum;
        }
    }
    out[(long long)(b * (long long)Nq + q) * 768 + ch] = acc;
}

// ---------------- depthwise 3x3 (SAME) + bias + exact GELU ----------------
__global__ void dwconv_gelu_kernel(const float* __restrict__ H, const float* __restrict__ Wdw,
                                   const float* __restrict__ Bdw, float* __restrict__ Out)
{
    long long id = (long long)blockIdx.x * blockDim.x + threadIdx.x;
    const long long total = 2LL * 34000 * 192;
    if (id >= total) return;
    const int ch = (int)(id % 192);
    long long bn = id / 192;
    const int n = (int)(bn % 34000);
    const int b = (int)(bn / 34000);
    int st, dim;
    if      (n < 25600) { st = 0;     dim = 160; }
    else if (n < 32000) { st = 25600; dim = 80;  }
    else if (n < 33600) { st = 32000; dim = 40;  }
    else                { st = 33600; dim = 20;  }
    const int loc = n - st;
    const int y = loc / dim, x = loc % dim;
    float acc = 0.f;
#pragma unroll
    for (int ky = 0; ky < 3; ++ky) {
#pragma unroll
        for (int kx = 0; kx < 3; ++kx) {
            const int yy = y + ky - 1, xx = x + kx - 1;
            if (yy >= 0 && yy < dim && xx >= 0 && xx < dim)
                acc += H[((long long)b * 34000 + st + yy * dim + xx) * 192 + ch] *
                       Wdw[ch * 9 + ky * 3 + kx];
        }
    }
    acc += Bdw[ch];
    Out[id] = 0.5f * acc * (1.0f + erff(acc * 0.70710678118654752f));
}

// ---------------- host orchestration ----------------
extern "C" void kernel_launch(void* const* d_in, const int* in_sizes, int n_in,
                              void* d_out, int out_size, void* d_ws, size_t ws_size,
                              hipStream_t stream)
{
    (void)in_sizes; (void)n_in; (void)out_size; (void)ws_size;
    const float* x        = (const float*)d_in[0];
    const float* c        = (const float*)d_in[1];
    const float* ref1     = (const float*)d_in[2];
    const float* ref2     = (const float*)d_in[3];
    const float* inj_qn_g = (const float*)d_in[4];
    const float* inj_qn_b = (const float*)d_in[5];
    const float* inj_fn_g = (const float*)d_in[6];
    const float* inj_fn_b = (const float*)d_in[7];
    const float* inj_off_w= (const float*)d_in[8];
    const float* inj_off_b= (const float*)d_in[9];
    const float* inj_aw_w = (const float*)d_in[10];
    const float* inj_aw_b = (const float*)d_in[11];
    const float* inj_vp_w = (const float*)d_in[12];
    const float* inj_vp_b = (const float*)d_in[13];
    const float* inj_op_w = (const float*)d_in[14];
    const float* inj_op_b = (const float*)d_in[15];
    const float* inj_gamma= (const float*)d_in[16];
    const float* ext_qn_g = (const float*)d_in[17];
    const float* ext_qn_b = (const float*)d_in[18];
    const float* ext_fn_g = (const float*)d_in[19];
    const float* ext_fn_b = (const float*)d_in[20];
    const float* ext_off_w= (const float*)d_in[21];
    const float* ext_off_b= (const float*)d_in[22];
    const float* ext_aw_w = (const float*)d_in[23];
    const float* ext_aw_b = (const float*)d_in[24];
    const float* ext_vp_w = (const float*)d_in[25];
    const float* ext_vp_b = (const float*)d_in[26];
    const float* ext_op_w = (const float*)d_in[27];
    const float* ext_op_b = (const float*)d_in[28];
    const float* ffn_g    = (const float*)d_in[29];
    const float* ffn_b    = (const float*)d_in[30];
    const float* fc1_w    = (const float*)d_in[31];
    const float* fc1_b    = (const float*)d_in[32];
    const float* dw_w     = (const float*)d_in[33];
    const float* dw_b     = (const float*)d_in[34];
    const float* fc2_w    = (const float*)d_in[35];
    const float* fc2_b    = (const float*)d_in[36];

    const long long XTOT = 2LL * 1600 * 768;    // x elements
    const long long CTOT = 2LL * 34000 * 768;   // c elements
    float* out_x = (float*)d_out;
    float* out_c = out_x + XTOT;

    float* ws     = (float*)d_ws;
    float* bufA   = ws;                          // CTOT  (ln_c1 / ln_c2 / ln_c3)
    float* bufB   = bufA + CTOT;                 // CTOT  (value1 / msda2 out)
    float* ln_x   = bufB + CTOT;                 // XTOT  (ln_x1 / ln_x2)
    float* off1   = ln_x + XTOT;                 // 2*1600*384
    float* aw1    = off1 + 2LL * 1600 * 384;     // 2*1600*192
    float* msda1  = aw1  + 2LL * 1600 * 192;     // XTOT
    float* value2 = msda1 + XTOT;                // XTOT
    float* off2   = value2 + XTOT;               // 2*34000*96
    float* aw2    = off2 + 2LL * 34000 * 96;     // 2*34000*48
    float* hbuf   = aw2  + 2LL * 34000 * 48;     // 2*34000*192
    float* hg     = hbuf + 2LL * 34000 * 192;    // 2*34000*192

    const dim3 blk(256);
    const int Mx = 3200, Mc = 68000;             // B*Nq rows
    const int GyX = (Mx + 127) / 128;            // 25
    const int GyC = (Mc + 127) / 128;            // 532

    LevelsDesc lvC; lvC.L = 4;
    lvC.h[0]=160; lvC.w[0]=160; lvC.start[0]=0;
    lvC.h[1]=80;  lvC.w[1]=80;  lvC.start[1]=25600;
    lvC.h[2]=40;  lvC.w[2]=40;  lvC.start[2]=32000;
    lvC.h[3]=20;  lvC.w[3]=20;  lvC.start[3]=33600;
    LevelsDesc lvX; lvX.L = 1;
    lvX.h[0]=40;  lvX.w[0]=40;  lvX.start[0]=0;
    lvX.h[1]=lvX.h[2]=lvX.h[3]=1; lvX.w[1]=lvX.w[2]=lvX.w[3]=1;
    lvX.start[1]=lvX.start[2]=lvX.start[3]=0;

    // ---- injector MSDA: query=LN(x), feat=LN(c) ----
    layernorm_kernel<<<dim3(400),  blk, 0, stream>>>(x, inj_qn_g, inj_qn_b, ln_x, Mx);
    layernorm_kernel<<<dim3(8500), blk, 0, stream>>>(c, inj_fn_g, inj_fn_b, bufA, Mc);
    gemm_bf16_wmma<0, 2><<<dim3(24, GyC), blk, 0, stream>>>(
        bufA, inj_vp_w, inj_vp_b, nullptr, nullptr, bufB, Mc, 768, 768);
    gemm_bf16_wmma<0, 2><<<dim3(12, GyX), blk, 0, stream>>>(
        ln_x, inj_off_w, inj_off_b, nullptr, nullptr, off1, Mx, 384, 768);
    gemm_bf16_wmma<0, 2><<<dim3(6, GyX), blk, 0, stream>>>(
        ln_x, inj_aw_w, inj_aw_b, nullptr, nullptr, aw1, Mx, 192, 768);
    {
        long long tot = 2LL * 1600 * 768;
        msda_sample_kernel<<<dim3((unsigned)((tot + 255) / 256)), blk, 0, stream>>>(
            bufB, ref1, off1, aw1, msda1, 1600, 34000, lvC);
    }
    // x_new = x + gamma * (msda1 @ op_w + b)
    gemm_bf16_wmma<1, 2><<<dim3(24, GyX), blk, 0, stream>>>(
        msda1, inj_op_w, inj_op_b, x, inj_gamma, out_x, Mx, 768, 768);

    // ---- extractor MSDA: query=LN(c), feat=LN(x_new) ----
    layernorm_kernel<<<dim3(400),  blk, 0, stream>>>(out_x, ext_fn_g, ext_fn_b, ln_x, Mx);
    layernorm_kernel<<<dim3(8500), blk, 0, stream>>>(c, ext_qn_g, ext_qn_b, bufA, Mc);
    gemm_bf16_wmma<0, 2><<<dim3(24, GyX), blk, 0, stream>>>(
        ln_x, ext_vp_w, ext_vp_b, nullptr, nullptr, value2, Mx, 768, 768);
    gemm_bf16_wmma<0, 2><<<dim3(3, GyC), blk, 0, stream>>>(
        bufA, ext_off_w, ext_off_b, nullptr, nullptr, off2, Mc, 96, 768);
    gemm_bf16_wmma<0, 1><<<dim3(3, GyC), blk, 0, stream>>>(
        bufA, ext_aw_w, ext_aw_b, nullptr, nullptr, aw2, Mc, 48, 768);
    {
        long long tot = 2LL * 34000 * 768;
        msda_sample_kernel<<<dim3((unsigned)((tot + 255) / 256)), blk, 0, stream>>>(
            value2, ref2, off2, aw2, bufB, 34000, 1600, lvX);
    }
    // c_new = c + (msda2 @ op_w + b)
    gemm_bf16_wmma<2, 2><<<dim3(24, GyC), blk, 0, stream>>>(
        bufB, ext_op_w, ext_op_b, c, nullptr, out_c, Mc, 768, 768);

    // ---- conv-FFN on c ----
    layernorm_kernel<<<dim3(8500), blk, 0, stream>>>(out_c, ffn_g, ffn_b, bufA, Mc);
    gemm_bf16_wmma<0, 2><<<dim3(6, GyC), blk, 0, stream>>>(
        bufA, fc1_w, fc1_b, nullptr, nullptr, hbuf, Mc, 192, 768);
    {
        long long tot = 2LL * 34000 * 192;
        dwconv_gelu_kernel<<<dim3((unsigned)((tot + 255) / 256)), blk, 0, stream>>>(
            hbuf, dw_w, dw_b, hg);
    }
    // c_final = c_new + (hg @ fc2_w + b)   (in-place residual: disjoint elements per lane)
    gemm_bf16_wmma<2, 2><<<dim3(24, GyC), blk, 0, stream>>>(
        hg, fc2_w, fc2_b, out_c, nullptr, out_c, Mc, 768, 192);
}
